// Encoder_755914244112
// MI455X (gfx1250) — matrix operand order repaired and checked by reading, not compiled, not run on hardware
//
#include <hip/hip_runtime.h>
#include <math.h>

// ---------------------------------------------------------------------------
// Encoder (attention-gated LSTM), B=256 T=128 N=512 M=128.
// Key algebraic fact: softmax(hs_score[:,None] + x_score) == softmax(x_score)
// (row-constant shift cancels), so alpha is time-invariant. This decouples
// the scan:  X_tilde = alpha * X  and  Gx = X_tilde @ W_ih^T  are hoisted
// into one bf16 WMMA GEMM; only h @ W_hh^T + LSTM cell stays sequential,
// and that part is independent per batch row (16 rows per block, t-loop
// inside the kernel, h kept as bf16 A-fragments in LDS).
// ---------------------------------------------------------------------------

typedef __attribute__((ext_vector_type(16))) __bf16 v16bf;
typedef __attribute__((ext_vector_type(8)))  __bf16 v8bf;
typedef __attribute__((ext_vector_type(4)))  __bf16 v4bf;
typedef __attribute__((ext_vector_type(8)))  float  v8f;
typedef __attribute__((ext_vector_type(4)))  float  v4f;

#define B_ 256
#define T_ 128
#define N_ 512
#define M_ 128
#define J_ 512 /* 4*M */

// A-fragment (16x32 bf16): lane<16 -> row=lane, K in {0..7,16..23};
// lane>=16 -> row=lane-16, K in {8..15,24..31}.  p0 points at (row0,k0).
__device__ __forceinline__ v16bf load_frag_a(const __bf16* __restrict__ p0,
                                             int ldk, int lane) {
  int r  = lane & 15;
  int kb = (lane >> 4) << 3;                 // 0 or 8
  const __bf16* p = p0 + (size_t)r * ldk + kb;
  v8bf lo = *(const v8bf*)(p);               // K = kb .. kb+7
  v8bf hi = *(const v8bf*)(p + 16);          // K = kb+16 .. kb+23
  v16bf a;
#pragma unroll
  for (int i = 0; i < 8; ++i) { a[i] = lo[i]; a[i + 8] = hi[i]; }
  return a;
}

// B-fragment (32x16 bf16): lane<16 -> col=lane, K in {0..7,16..23};
// lane>=16 -> col=lane-16, K in {8..15,24..31}.  B[k][n] = W[j0+n][k], W row-major
// with leading dim ldk over k (contraction along contiguous axis -> b128s).
__device__ __forceinline__ v16bf load_frag_b(const __bf16* __restrict__ w,
                                             int j0, int k0, int ldk, int lane) {
  int n  = lane & 15;
  int kb = k0 + ((lane >> 4) << 3);          // +0 or +8
  const __bf16* p = w + (size_t)(j0 + n) * ldk + kb;
  v8bf lo = *(const v8bf*)(p);
  v8bf hi = *(const v8bf*)(p + 16);
  v16bf b;
#pragma unroll
  for (int i = 0; i < 8; ++i) { b[i] = lo[i]; b[i + 8] = hi[i]; }
  return b;
}

__device__ __forceinline__ v8f wmma_bf16(v16bf a, v16bf b, v8f c) {
  return __builtin_amdgcn_wmma_f32_16x16x32_bf16(false, a, false, b,
                                                 (short)0, c, false, false);
}

// ---------------------------------------------------------------------------
// K0: f32 -> bf16 weight conversion
// ---------------------------------------------------------------------------
__global__ void convert_weights_kernel(const float* __restrict__ Wih,
                                       const float* __restrict__ Whh,
                                       __bf16* __restrict__ Wih_b,
                                       __bf16* __restrict__ Whh_b) {
  int idx = blockIdx.x * blockDim.x + threadIdx.x;
  if (idx < J_ * N_) Wih_b[idx] = (__bf16)Wih[idx];
  if (idx < J_ * M_) Whh_b[idx] = (__bf16)Whh[idx];
}

// ---------------------------------------------------------------------------
// K1: per batch row b: scores s[n] = sum_t X[b,t,n]*w_x[t]; alpha = softmax_n;
//     X_tilde = alpha * X  (written f32 to d_out and bf16 to workspace).
// ---------------------------------------------------------------------------
__global__ __launch_bounds__(256)
void attn_scale_kernel(const float* __restrict__ X,
                       const float* __restrict__ Wattn,
                       float* __restrict__ Xt_out,
                       __bf16* __restrict__ Xt_bf) {
  const int b   = blockIdx.x;
  const int tid = threadIdx.x;
  __shared__ __align__(16) float wx[T_];
  __shared__ __align__(16) float sc[N_];
  __shared__ float red[16];
  if (tid < T_) wx[tid] = Wattn[2 * M_ + tid];   // w_x slice of W_attn
  __syncthreads();

  const float* Xb = X + (size_t)b * T_ * N_;
  float s0 = 0.f, s1 = 0.f;
  for (int t = 0; t < T_; ++t) {
    float w = wx[t];
    s0 = fmaf(Xb[t * N_ + tid],       w, s0);
    s1 = fmaf(Xb[t * N_ + tid + 256], w, s1);
  }
  // block max over 512 scores
  float lm = fmaxf(s0, s1);
#pragma unroll
  for (int off = 16; off >= 1; off >>= 1) lm = fmaxf(lm, __shfl_xor(lm, off, 32));
  if ((tid & 31) == 0) red[tid >> 5] = lm;
  __syncthreads();
  if (tid == 0) {
    float m = red[0];
    for (int i = 1; i < 8; ++i) m = fmaxf(m, red[i]);
    red[8] = m;
  }
  __syncthreads();
  float mx = red[8];
  float e0 = __expf(s0 - mx), e1 = __expf(s1 - mx);
  sc[tid] = e0; sc[tid + 256] = e1;
  float ls = e0 + e1;
#pragma unroll
  for (int off = 16; off >= 1; off >>= 1) ls += __shfl_xor(ls, off, 32);
  if ((tid & 31) == 0) red[tid >> 5] = ls;
  __syncthreads();
  if (tid == 0) {
    float s = 0.f;
    for (int i = 0; i < 8; ++i) s += red[i];
    red[9] = 1.0f / s;
  }
  __syncthreads();
  const float inv = red[9];

  // scale: T*N = 65536 elements as float4 (alpha repeats every 512 along n)
  const v4f* Xv = (const v4f*)Xb;
  v4f*  Ov = (v4f*)(Xt_out + (size_t)b * T_ * N_);
  v4bf* Bv = (v4bf*)(Xt_bf + (size_t)b * T_ * N_);
  for (int it = 0; it < (T_ * N_ / 4) / 256; ++it) {
    int i4 = it * 256 + tid;
    int n  = (i4 * 4) & (N_ - 1);
    v4f xv = Xv[i4];
    v4f av = *(const v4f*)&sc[n];
    v4f o; v4bf ob;
#pragma unroll
    for (int j = 0; j < 4; ++j) { o[j] = xv[j] * (av[j] * inv); ob[j] = (__bf16)o[j]; }
    Ov[i4] = o;
    Bv[i4] = ob;
  }
}

// ---------------------------------------------------------------------------
// K2: Gx(32768 x 512) = X_tilde(bf16) @ W_ih^T(bf16), f32 accumulate.
//     Per wave: 32 rows x 64 cols (2 A-frags x 4 B-frags, 8 accumulators).
//     Block = 8 waves (4 row-groups x 2 col-groups) -> 128x128 tile.
// ---------------------------------------------------------------------------
__global__ __launch_bounds__(256)
void gemm_xw_kernel(const __bf16* __restrict__ Xt,
                    const __bf16* __restrict__ Wih,
                    float* __restrict__ Gx) {
  const int lane = threadIdx.x & 31;
  const int wave = threadIdx.x >> 5;
  const int row0 = blockIdx.x * 128 + (wave & 3) * 32;
  const int col0 = blockIdx.y * 128 + (wave >> 2) * 64;

  v8f acc[2][4];
  const v8f vz = {0.f, 0.f, 0.f, 0.f, 0.f, 0.f, 0.f, 0.f};
#pragma unroll
  for (int h = 0; h < 2; ++h)
#pragma unroll
    for (int q = 0; q < 4; ++q) acc[h][q] = vz;

  for (int k0 = 0; k0 < N_; k0 += 32) {
    v16bf a0 = load_frag_a(Xt + (size_t)row0 * N_ + k0, N_, lane);
    v16bf a1 = load_frag_a(Xt + (size_t)(row0 + 16) * N_ + k0, N_, lane);
#pragma unroll
    for (int q = 0; q < 4; ++q) {
      v16bf bq = load_frag_b(Wih, col0 + q * 16, k0, N_, lane);
      acc[0][q] = wmma_bf16(a0, bq, acc[0][q]);
      acc[1][q] = wmma_bf16(a1, bq, acc[1][q]);
    }
  }
  const int rlo = (lane >> 4) << 3;  // D row offset: 0 / 8
  const int cn  = lane & 15;
#pragma unroll
  for (int h = 0; h < 2; ++h)
#pragma unroll
    for (int q = 0; q < 4; ++q)
#pragma unroll
      for (int r = 0; r < 8; ++r)
        Gx[(size_t)(row0 + h * 16 + rlo + r) * J_ + (col0 + q * 16 + cn)]
            = acc[h][q][r];
}

// ---------------------------------------------------------------------------
// K3: sequential LSTM. Each block owns 16 batch rows; loops all T steps.
//     Per step: gates_h(16x512) = h(16x128,bf16 in LDS) @ W_hh^T via WMMA
//     (each of 8 waves: 4 j-tiles x 4 K-chunks), then elementwise cell with
//     c in registers, h written back to LDS as bf16 A-fragments.
// ---------------------------------------------------------------------------
__global__ __launch_bounds__(256)
void lstm_kernel(const float* __restrict__ Gx,
                 const __bf16* __restrict__ Whh_b,
                 const float* __restrict__ b_ih,
                 const float* __restrict__ b_hh,
                 float* __restrict__ Henc) {
  const int tid  = threadIdx.x;
  const int lane = tid & 31;
  const int wave = tid >> 5;
  const int b0   = blockIdx.x * 16;

  __shared__ __align__(16) __bf16 h_lds[16][136]; // pad: stride 272B, conflict-free
  __shared__ float gates[16][516];                // pad kills half-wave conflicts
  __shared__ float bias[J_];

  for (int i = tid; i < 16 * 136; i += 256) ((__bf16*)h_lds)[i] = (__bf16)0.0f;
  for (int i = tid; i < J_; i += 256) bias[i] = b_ih[i] + b_hh[i];
  float c_reg[8];
#pragma unroll
  for (int i = 0; i < 8; ++i) c_reg[i] = 0.0f;
  __syncthreads();

  const int arow = lane & 15;
  const int akb  = (lane >> 4) << 3;
  const int rlo  = (lane >> 4) << 3;
  const int cn   = lane & 15;
  const v8f vz = {0.f, 0.f, 0.f, 0.f, 0.f, 0.f, 0.f, 0.f};

  for (int t = 0; t < T_; ++t) {
    // ---- phase 1: gates_h = h @ W_hh^T (K = 128) ----
    v8f acc[4];
#pragma unroll
    for (int q = 0; q < 4; ++q) acc[q] = vz;
#pragma unroll
    for (int kc = 0; kc < 4; ++kc) {
      const int k0 = kc * 32;
      const __bf16* hp = &h_lds[arow][k0 + akb];
      v8bf lo = *(const v8bf*)hp;
      v8bf hi = *(const v8bf*)(hp + 16);
      v16bf a;
#pragma unroll
      for (int i = 0; i < 8; ++i) { a[i] = lo[i]; a[i + 8] = hi[i]; }
#pragma unroll
      for (int q = 0; q < 4; ++q) {
        const int j0 = (wave * 4 + q) * 16;
        v16bf bq = load_frag_b(Whh_b, j0, k0, M_, lane);
        acc[q] = wmma_bf16(a, bq, acc[q]);
      }
    }
#pragma unroll
    for (int q = 0; q < 4; ++q) {
      const int j0 = (wave * 4 + q) * 16;
#pragma unroll
      for (int r = 0; r < 8; ++r) gates[rlo + r][j0 + cn] = acc[q][r];
    }
    __syncthreads();

    // ---- phase 2: LSTM cell (2048 cells, 8 per thread) ----
#pragma unroll
    for (int i = 0; i < 8; ++i) {
      const int cid = tid + 256 * i;
      const int bb  = cid >> 7;          // 0..15
      const int m   = cid & (M_ - 1);    // 0..127
      const size_t gxb = ((size_t)(b0 + bb) * T_ + t) * (size_t)J_;
      float gi = gates[bb][m]       + Gx[gxb + m]       + bias[m];
      float gf = gates[bb][m + 128] + Gx[gxb + m + 128] + bias[m + 128];
      float gg = gates[bb][m + 256] + Gx[gxb + m + 256] + bias[m + 256];
      float go = gates[bb][m + 384] + Gx[gxb + m + 384] + bias[m + 384];
      float si = 1.0f / (1.0f + __expf(-gi));
      float sf = 1.0f / (1.0f + __expf(-gf));
      float so = 1.0f / (1.0f + __expf(-go));
      float cn2 = sf * c_reg[i] + si * tanhf(gg);
      float hn  = so * tanhf(cn2);
      c_reg[i] = cn2;
      Henc[((size_t)(b0 + bb) * T_ + t) * M_ + m] = hn;
      h_lds[bb][m] = (__bf16)hn;
    }
    __syncthreads();
  }
}

// ---------------------------------------------------------------------------
extern "C" void kernel_launch(void* const* d_in, const int* in_sizes, int n_in,
                              void* d_out, int out_size, void* d_ws, size_t ws_size,
                              hipStream_t stream) {
  (void)in_sizes; (void)n_in; (void)out_size;
  const float* X      = (const float*)d_in[0];
  const float* W_attn = (const float*)d_in[1];
  // d_in[2] = b_attn: row-constant -> cancels inside softmax, unused.
  const float* W_ih   = (const float*)d_in[3];
  const float* W_hh   = (const float*)d_in[4];
  const float* b_ih   = (const float*)d_in[5];
  const float* b_hh   = (const float*)d_in[6];

  float* Xt_out = (float*)d_out;                          // (B,T,N) f32
  float* Henc   = (float*)d_out + (size_t)B_ * T_ * N_;   // (B,T,M) f32

  // workspace: 512K (Wih bf16) + 128K (Whh bf16) + 32M (Xt bf16) + 64M (Gx f32)
  char* ws = (char*)d_ws;
  __bf16* Wih_b = (__bf16*)ws;
  __bf16* Whh_b = (__bf16*)(ws + (512u << 10));
  __bf16* Xt_bf = (__bf16*)(ws + (640u << 10));
  float*  Gx    = (float*)(ws + (640u << 10) + (size_t)B_ * T_ * N_ * 2);
  (void)ws_size; // requires ~101 MB

  convert_weights_kernel<<<(J_ * N_ + 255) / 256, 256, 0, stream>>>(
      W_ih, W_hh, Wih_b, Whh_b);
  attn_scale_kernel<<<B_, 256, 0, stream>>>(X, W_attn, Xt_out, Xt_bf);
  dim3 g2(B_ * T_ / 128, J_ / 128);
  gemm_xw_kernel<<<g2, 256, 0, stream>>>(Xt_bf, Wih_b, Gx);
  lstm_kernel<<<B_ / 16, 256, 0, stream>>>(Gx, Whh_b, b_ih, b_hh, Henc);
}
